// AttentionPairBias_7026566497091
// MI455X (gfx1250) — matrix operand order repaired and checked by reading, not compiled, not run on hardware
//
#include <hip/hip_runtime.h>
#include <hip/hip_bf16.h>

// ---------------------------------------------------------------------------
// AttentionPairBias for MI455X (gfx1250, wave32, WMMA bf16 16x16x32 + TDM)
// B=1, N=768 tokens, DS=384, DP=128, H=16, head dim 24 (padded to 32)
// ---------------------------------------------------------------------------

typedef __bf16 bf16;
typedef __attribute__((ext_vector_type(16))) __bf16 v16bf;
typedef __attribute__((ext_vector_type(8)))  __bf16 v8bf;
typedef __attribute__((ext_vector_type(8)))  float  v8f;
typedef __attribute__((ext_vector_type(4)))  float  v4f;
typedef __attribute__((ext_vector_type(4)))  unsigned int u32x4;
typedef __attribute__((ext_vector_type(8)))  int    i32x8;
typedef __attribute__((ext_vector_type(4)))  int    i32x4;

#define N_TOK 768
#define DS    384
#define DP    128
#define H     16
#define DH    24
#define DHP   32
#define LN_EPS 1e-5f
#define PK_CHUNKS 4            // 16-row tiles per wave in pair kernel

// ---- workspace layout (bytes) ---------------------------------------------
static constexpr size_t OFF_SX   = 0;                                   // bf16 768*384
static constexpr size_t OFF_WQT  = OFF_SX  + (size_t)N_TOK*DS*2;        // bf16 384*384 (transposed)
static constexpr size_t OFF_WKT  = OFF_WQT + (size_t)DS*DS*2;
static constexpr size_t OFF_WVT  = OFF_WKT + (size_t)DS*DS*2;
static constexpr size_t OFF_WOT  = OFF_WVT + (size_t)DS*DS*2;
static constexpr size_t OFF_WBT  = OFF_WOT + (size_t)DS*DS*2;           // bf16 16*128 (transposed)
static constexpr size_t OFF_QP   = OFF_WBT + (size_t)H*DP*2;            // bf16 768*16*32 padded
static constexpr size_t OFF_KP   = OFF_QP  + (size_t)N_TOK*H*DHP*2;
static constexpr size_t OFF_VT   = OFF_KP  + (size_t)N_TOK*H*DHP*2;     // bf16 [H][32][768]
static constexpr size_t OFF_ATT  = OFF_VT  + (size_t)H*DHP*N_TOK*2;     // bf16 768*384
static constexpr size_t OFF_BIAS = OFF_ATT + (size_t)N_TOK*DS*2;        // f32  [H][768][768]

// ---- WMMA operand loaders --------------------------------------------------
// A operand (16x32 bf16): lane m = lane&15 holds row m; hi half of wave holds
// K = {8..15, 24..31}, lo half holds K = {0..7, 16..23}.
static __device__ inline v16bf load_a(const bf16* rowk, int hi) {
  v8bf lo = *(const v8bf*)(rowk + hi * 8);
  v8bf hh = *(const v8bf*)(rowk + 16 + hi * 8);
  v16bf r;
#pragma unroll
  for (int i = 0; i < 8; ++i) { r[i] = lo[i]; r[i + 8] = hh[i]; }
  return r;
}

// B operand (32x16 bf16): lane n = lane&15 holds column n; lanes 0-15 hold
// K = {0..7, 16..23}, lanes 16-31 hold K = {8..15, 24..31} (matches A).
static __device__ inline v16bf load_b(const bf16* colk, int hi) {
  v8bf lo = *(const v8bf*)(colk + hi * 8);
  v8bf hh = *(const v8bf*)(colk + 16 + hi * 8);
  v16bf r;
#pragma unroll
  for (int i = 0; i < 8; ++i) { r[i] = lo[i]; r[i + 8] = hh[i]; }
  return r;
}

static __device__ inline v8f wmma_bf16(v16bf a, v16bf b, v8f c) {
  return __builtin_amdgcn_wmma_f32_16x16x32_bf16(false, a, false, b,
                                                 (short)0, c, false, false);
}

// ---- Tensor Data Mover: 1-D tile (2048 x 4B = 8KB) global -> LDS -----------
// LDS byte offset of a __shared__ object = low 32 bits of its generic address
// (addrspace(3) occupies addr[31:0] of the flat shared aperture).
static __device__ inline unsigned lds_off(const void* p) {
  return (unsigned)(unsigned long long)p;
}

static __device__ inline void tdm_load_8k(unsigned lds_byte,
                                          unsigned long long gaddr) {
  const unsigned elems = 16 * DP;                 // 2048 elements of 4 bytes
  u32x4 g0;
  g0[0] = 1u;                                     // count=1, user descriptor
  g0[1] = lds_byte;                               // lds_addr [63:32]
  g0[2] = (unsigned)gaddr;                        // global_addr [95:64]
  g0[3] = (unsigned)((gaddr >> 32) & 0x1FFFFFFu)  // global_addr [120:96]
        | (2u << 30);                             // type=2 ("image") [127:126]
  i32x8 g1;
  g1[0] = (int)(2u << 16);                        // wg_mask=0, data_size=2 (4B)
  g1[1] = (int)((elems & 0xFFFFu) << 16);         // tensor_dim0[15:0] @ [63:48]
  g1[2] = (int)((elems >> 16) | (1u << 16));      // tensor_dim0[31:16], tensor_dim1=1
  g1[3] = (int)((elems & 0xFFFFu) << 16);         // tile_dim0 @ [127:112]
  g1[4] = 0;                                      // tile_dim1=0 (1-D), tile_dim2=0
  g1[5] = (int)elems;                             // tensor_dim0_stride[31:0]
  g1[6] = 0;                                      // stride hi, stride1 lo
  g1[7] = 0;                                      // stride1 hi
  i32x4 z4 = {0, 0, 0, 0};
#if __has_include(<hip/amd_detail/amd_gfx1250_TDM.h>)
  i32x8 z8 = {0, 0, 0, 0, 0, 0, 0, 0};
  __builtin_amdgcn_tensor_load_to_lds(g0, g1, z4, z4, z8, 0);
#else
  __builtin_amdgcn_tensor_load_to_lds(g0, g1, z4, z4, 0);
#endif
}

// ---------------------------------------------------------------------------
// prep: fp32->bf16 converts, weight transposes, zero-pad Q/K/Vt buffers
// ---------------------------------------------------------------------------
__global__ void prep_kernel(const float* __restrict__ single,
                            const float* __restrict__ Wq,
                            const float* __restrict__ Wk,
                            const float* __restrict__ Wv,
                            const float* __restrict__ Wo,
                            const float* __restrict__ Wb,
                            bf16* sx, bf16* wqt, bf16* wkt, bf16* wvt,
                            bf16* wot, bf16* wbt, bf16* qp, bf16* kp, bf16* vt) {
  int i = blockIdx.x * blockDim.x + threadIdx.x;
  if (i < N_TOK * DS) sx[i] = (bf16)single[i];
  if (i < DS * DS) {
    int k = i / DS, n = i % DS;          // W stored [in][out]; build [out][in]
    wqt[n * DS + k] = (bf16)Wq[i];
    wkt[n * DS + k] = (bf16)Wk[i];
    wvt[n * DS + k] = (bf16)Wv[i];
    wot[n * DS + k] = (bf16)Wo[i];
  }
  if (i < DP * H) {
    int dp = i / H, hh = i % H;
    wbt[hh * DP + dp] = (bf16)Wb[i];
  }
  if (i < N_TOK * H * DHP) {             // zero padding lanes d=24..31
    qp[i] = (bf16)0.0f; kp[i] = (bf16)0.0f; vt[i] = (bf16)0.0f;
  }
}

// ---------------------------------------------------------------------------
// generic 16x16-tile bf16 GEMM: C[MxN] = A[MxK] * B[KxN], B given transposed.
// mode 0: store bf16 into padded [tok][H][DHP]   (Q, K)
// mode 1: store bf16 into [H][DHP][tok]          (V transposed)
// mode 2: store f32 row-major                    (final output)
// ---------------------------------------------------------------------------
__global__ void gemm_kernel(const bf16* __restrict__ A,
                            const bf16* __restrict__ Bt,
                            int M, int Ncols, int K, int mode, void* outp) {
  int lane = threadIdx.x & 31;
  int wave = threadIdx.x >> 5;
  int tile = blockIdx.x * 4 + wave;
  int tiles_n = Ncols / 16;
  int m0 = (tile / tiles_n) * 16;
  int n0 = (tile % tiles_n) * 16;
  int lm = lane & 15, hi = lane >> 4;

  v8f acc = {};
  const bf16* arow = A  + (size_t)(m0 + lm) * K;
  const bf16* brow = Bt + (size_t)(n0 + lm) * K;
  for (int k = 0; k < K; k += 32) {
    v16bf a = load_a(arow + k, hi);
    v16bf b = load_b(brow + k, hi);
    acc = wmma_bf16(a, b, acc);
  }

  int ncol = n0 + lm;
  if (mode == 2) {
    float* o = (float*)outp;
#pragma unroll
    for (int r = 0; r < 8; ++r)
      o[(size_t)(m0 + r + 8 * hi) * Ncols + ncol] = acc[r];
  } else if (mode == 0) {
    bf16* o = (bf16*)outp;
    int h = ncol / DH, d = ncol % DH;
#pragma unroll
    for (int r = 0; r < 8; ++r)
      o[(size_t)(m0 + r + 8 * hi) * (H * DHP) + h * DHP + d] = (bf16)acc[r];
  } else {
    bf16* o = (bf16*)outp;
    int h = ncol / DH, d = ncol % DH;
#pragma unroll
    for (int r = 0; r < 8; ++r)
      o[((size_t)h * DHP + d) * N_TOK + (m0 + r + 8 * hi)] = (bf16)acc[r];
  }
}

// ---------------------------------------------------------------------------
// pair LayerNorm + Wb projection, fused.  One wave handles PK_CHUNKS tiles of
// 16 flattened (q,k) rows; pair tiles are DMA'd into LDS by the Tensor Data
// Mover, double-buffered on TENSORcnt.  Bias stored head-major [H][q][k] f32.
// ---------------------------------------------------------------------------
__global__ void pair_kernel(const float* __restrict__ pair,
                            const float* __restrict__ gamma,
                            const float* __restrict__ beta,
                            const bf16* __restrict__ wbt,
                            float* __restrict__ bias) {
  __shared__ float stage[4][2][16 * DP];   // 4 waves x double buffer x 8KB
  __shared__ bf16  nrm[4][16 * DP];        // normalized bf16 tile per wave
  int lane = threadIdx.x & 31;
  int wave = threadIdx.x >> 5;
  int lm = lane & 15, hi = lane >> 4;
  size_t tile0 = ((size_t)blockIdx.x * 4 + wave) * PK_CHUNKS; // 16-row tiles

  v4f g  = *(const v4f*)(gamma + lane * 4);
  v4f bt = *(const v4f*)(beta  + lane * 4);
  float* st0 = stage[wave][0];
  float* st1 = stage[wave][1];
  bf16*  myn = nrm[wave];
  unsigned long long pbase = (unsigned long long)pair;

  // prefetch tile 0 via TDM
  tdm_load_8k(lds_off(st0), pbase + tile0 * (16 * DP * 4ull));

#pragma unroll
  for (int c = 0; c < PK_CHUNKS; ++c) {
    float* st = (c & 1) ? st1 : st0;
    if (c + 1 < PK_CHUNKS) {
      tdm_load_8k(lds_off((c & 1) ? st0 : st1),
                  pbase + (tile0 + c + 1) * (16 * DP * 4ull));
      __builtin_amdgcn_s_wait_tensorcnt(1);   // tile c resident, c+1 in flight
    } else {
      __builtin_amdgcn_s_wait_tensorcnt(0);
    }
    asm volatile("" ::: "memory");

    // LayerNorm: 16 rows, 4 values/lane, parallel sum & sum-of-squares
    for (int r = 0; r < 16; ++r) {
      v4f x = *(const v4f*)(st + r * DP + lane * 4);
      float s  = x[0] + x[1] + x[2] + x[3];
      float sq = x[0]*x[0] + x[1]*x[1] + x[2]*x[2] + x[3]*x[3];
#pragma unroll
      for (int m = 1; m < 32; m <<= 1) {
        s  += __shfl_xor(s,  m, 32);
        sq += __shfl_xor(sq, m, 32);
      }
      float mu   = s * (1.0f / DP);
      float var  = fmaxf(sq * (1.0f / DP) - mu * mu, 0.0f);
      float rstd = rsqrtf(var + LN_EPS);
#pragma unroll
      for (int j = 0; j < 4; ++j)
        myn[r * DP + lane * 4 + j] = (bf16)((x[j] - mu) * rstd * g[j] + bt[j]);
    }
    __syncthreads();

    // 16x16 = (16x128) x (128x16) via 4 bf16 WMMAs
    v8f acc = {};
    const bf16* arow = myn + lm * DP;
    const bf16* brow = wbt + lm * DP;
#pragma unroll
    for (int k = 0; k < DP; k += 32) {
      v16bf a = load_a(arow + k, hi);
      v16bf b = load_b(brow + k, hi);
      acc = wmma_bf16(a, b, acc);
    }

    size_t base = (tile0 + c) * 16;        // flattened pair row of this tile
    int q  = (int)(base / N_TOK);
    int kb = (int)(base % N_TOK);          // 16-row chunk never crosses a q row
    size_t hb = (size_t)lm * ((size_t)N_TOK * N_TOK);
#pragma unroll
    for (int r = 0; r < 8; ++r)
      bias[hb + (size_t)q * N_TOK + kb + r + 8 * hi] = acc[r];
    __syncthreads();
  }
}

// ---------------------------------------------------------------------------
// flash attention: one wave per (16-query tile, head); online softmax over k
// in chunks of 32; scores and P*V via WMMA; P transposed through LDS.
// ---------------------------------------------------------------------------
__global__ void attn_kernel(const bf16* __restrict__ qp,
                            const bf16* __restrict__ kp,
                            const bf16* __restrict__ vt,
                            const float* __restrict__ bias,
                            bf16* __restrict__ attn_out) {
  __shared__ bf16 ldsp[16 * 32];
  int lane = threadIdx.x & 31;
  int lm = lane & 15, hi = lane >> 4;
  int h  = blockIdx.x & 15;
  int q0 = (blockIdx.x >> 4) * 16;
  const float scaling = 0.051031036307982884f;   // 384 ** -0.5 (full DS per ref)

  v16bf aq = load_a(qp + (size_t)(q0 + lm) * (H * DHP) + h * DHP, hi);
  v8f acc0 = {}, acc1 = {};
  float mrow[8], lrow[8];
#pragma unroll
  for (int r = 0; r < 8; ++r) { mrow[r] = -1e30f; lrow[r] = 0.0f; }

  const float* bb = bias + (size_t)h * ((size_t)N_TOK * N_TOK) + (size_t)q0 * N_TOK;

  for (int k0 = 0; k0 < N_TOK; k0 += 32) {
    // scores: Q(16x32) x K^T(32x16), two k-sub-tiles of 16
    v16bf bk0 = load_b(kp + (size_t)(k0 + lm) * (H * DHP) + h * DHP, hi);
    v16bf bk1 = load_b(kp + (size_t)(k0 + 16 + lm) * (H * DHP) + h * DHP, hi);
    v8f z = {};
    v8f s0 = wmma_bf16(aq, bk0, z);
    v8f s1 = wmma_bf16(aq, bk1, z);

#pragma unroll
    for (int r = 0; r < 8; ++r) {
      int row = r + 8 * hi;
      s0[r] = s0[r] * scaling + bb[(size_t)row * N_TOK + k0 + lm];
      s1[r] = s1[r] * scaling + bb[(size_t)row * N_TOK + k0 + 16 + lm];
      float mx = fmaxf(s0[r], s1[r]);
#pragma unroll
      for (int m = 1; m < 16; m <<= 1) mx = fmaxf(mx, __shfl_xor(mx, m, 32));
      float mn = fmaxf(mrow[r], mx);
      float sc = __expf(mrow[r] - mn);
      float p0 = __expf(s0[r] - mn);
      float p1 = __expf(s1[r] - mn);
      float rs = p0 + p1;
#pragma unroll
      for (int m = 1; m < 16; m <<= 1) rs += __shfl_xor(rs, m, 32);
      lrow[r] = lrow[r] * sc + rs;
      mrow[r] = mn;
      acc0[r] *= sc; acc1[r] *= sc;
      ldsp[row * 32 + lm]      = (bf16)p0;
      ldsp[row * 32 + 16 + lm] = (bf16)p1;
    }
    __syncthreads();

    // O += P(16x32) x V(32x16) for d=0..15 and d=16..31 (zero-padded past 24)
    v16bf ap  = load_a(ldsp + lm * 32, hi);
    v16bf bv0 = load_b(vt + ((size_t)h * DHP + lm) * N_TOK + k0, hi);
    v16bf bv1 = load_b(vt + ((size_t)h * DHP + 16 + lm) * N_TOK + k0, hi);
    acc0 = wmma_bf16(ap, bv0, acc0);
    acc1 = wmma_bf16(ap, bv1, acc1);
    __syncthreads();
  }

#pragma unroll
  for (int r = 0; r < 8; ++r) {
    float inv = 1.0f / lrow[r];
    int tok = q0 + r + 8 * hi;
    attn_out[(size_t)tok * DS + h * DH + lm] = (bf16)(acc0[r] * inv);
    if (lm < 8)
      attn_out[(size_t)tok * DS + h * DH + 16 + lm] = (bf16)(acc1[r] * inv);
  }
}

// ---------------------------------------------------------------------------
extern "C" void kernel_launch(void* const* d_in, const int* in_sizes, int n_in,
                              void* d_out, int out_size, void* d_ws, size_t ws_size,
                              hipStream_t stream) {
  (void)in_sizes; (void)n_in; (void)out_size; (void)ws_size;
  const float* single = (const float*)d_in[0];
  const float* pair   = (const float*)d_in[1];
  const float* Wq     = (const float*)d_in[2];
  const float* Wk     = (const float*)d_in[3];
  const float* Wv     = (const float*)d_in[4];
  const float* gamma  = (const float*)d_in[5];
  const float* beta   = (const float*)d_in[6];
  const float* Wb     = (const float*)d_in[7];
  const float* Wo     = (const float*)d_in[8];

  char* ws = (char*)d_ws;
  bf16*  sx   = (bf16*)(ws + OFF_SX);
  bf16*  wqt  = (bf16*)(ws + OFF_WQT);
  bf16*  wkt  = (bf16*)(ws + OFF_WKT);
  bf16*  wvt  = (bf16*)(ws + OFF_WVT);
  bf16*  wot  = (bf16*)(ws + OFF_WOT);
  bf16*  wbt  = (bf16*)(ws + OFF_WBT);
  bf16*  qp   = (bf16*)(ws + OFF_QP);
  bf16*  kp   = (bf16*)(ws + OFF_KP);
  bf16*  vtm  = (bf16*)(ws + OFF_VT);
  bf16*  attn = (bf16*)(ws + OFF_ATT);
  float* biasp= (float*)(ws + OFF_BIAS);

  // 1) converts / transposes / zero padding (covers max task = 393216 elems)
  prep_kernel<<<1536, 256, 0, stream>>>(single, Wq, Wk, Wv, Wo, Wb,
                                        sx, wqt, wkt, wvt, wot, wbt, qp, kp, vtm);

  // 2-4) Q/K/V projections: 768x384x384 bf16 WMMA GEMMs
  int tiles = (N_TOK / 16) * (DS / 16);   // 1152 tiles, 4 waves/block
  gemm_kernel<<<tiles / 4, 128, 0, stream>>>(sx, wqt, N_TOK, DS, DS, 0, qp);
  gemm_kernel<<<tiles / 4, 128, 0, stream>>>(sx, wkt, N_TOK, DS, DS, 0, kp);
  gemm_kernel<<<tiles / 4, 128, 0, stream>>>(sx, wvt, N_TOK, DS, DS, 1, vtm);

  // 5) fused LayerNorm + pair-bias projection; TDM streams the 302 MB tensor
  int pk_waves  = (N_TOK * N_TOK / 16) / PK_CHUNKS;   // 9216 waves
  pair_kernel<<<pk_waves / 4, 128, 0, stream>>>(pair, gamma, beta, wbt, biasp);

  // 6) flash attention with pair bias
  attn_kernel<<<(N_TOK / 16) * H, 32, 0, stream>>>(qp, kp, vtm, biasp, attn);

  // 7) output projection -> f32 d_out
  gemm_kernel<<<tiles / 4, 128, 0, stream>>>(attn, wot, N_TOK, DS, DS, 2,
                                             (float*)d_out);
}
